// MultiViewGNN_19851338842499
// MI455X (gfx1250) — matrix operand order repaired and checked by reading, not compiled, not run on hardware
//
#include <hip/hip_runtime.h>
#include <hip/hip_bf16.h>
#include <math.h>

typedef __attribute__((ext_vector_type(16))) _Float16 v16h;
typedef __attribute__((ext_vector_type(8)))  float    v8f;

#define D_FEAT 128

// ---------------------------------------------------------------- utilities

__global__ void zero_f32(float* __restrict__ p, long n) {
  long i = (long)blockIdx.x * blockDim.x + threadIdx.x;
  if (i < n) p[i] = 0.0f;
}

__global__ void vec_add(const float* __restrict__ a, const float* __restrict__ b,
                        float* __restrict__ o, long n) {
  long i = (long)blockIdx.x * blockDim.x + threadIdx.x;
  if (i < n) o[i] = a[i] + b[i];
}

// deg[dst] += w  (symmetric-norm degree; dst-side segment sum of edge weights)
__global__ void deg_accum(const int* __restrict__ dst, const float* __restrict__ w,
                          float* __restrict__ deg, int E) {
  int e = blockIdx.x * blockDim.x + threadIdx.x;
  if (e < E) atomicAdd(&deg[dst[e]], w[e]);
}

__global__ void deg_to_dinv(float* __restrict__ deg, int n) {
  int i = blockIdx.x * blockDim.x + threadIdx.x;
  if (i < n) {
    float v = deg[i];
    deg[i] = (v > 0.0f) ? (1.0f / sqrtf(v)) : 0.0f;
  }
}

// One wave per edge: lane c covers features [4c, 4c+4).
// out[dst] += h[src] * (dinv[src]*w*dinv[dst]); atomics resolve in L2.
__global__ void edge_scatter(const float* __restrict__ h,
                             const int* __restrict__ src,
                             const int* __restrict__ dst,
                             const float* __restrict__ w,
                             const float* __restrict__ dinv,
                             float* __restrict__ out, int E) {
  long t = (long)blockIdx.x * blockDim.x + threadIdx.x;
  int e = (int)(t >> 5);
  int c = (int)(t & 31);
  if (e >= E) return;
  int s = src[e], d = dst[e];
  float norm = dinv[s] * w[e] * dinv[d];
  const float4 hv = *reinterpret_cast<const float4*>(h + (long)s * D_FEAT + c * 4);
  float* op = out + (long)d * D_FEAT + c * 4;
  atomicAdd(op + 0, hv.x * norm);
  atomicAdd(op + 1, hv.y * norm);
  atomicAdd(op + 2, hv.z * norm);
  atomicAdd(op + 3, hv.w * norm);
}

// x = relu(x + b[f]) over [n, 128]
__global__ void bias_relu(float* __restrict__ x, const float* __restrict__ b, long total) {
  long i = (long)blockIdx.x * blockDim.x + threadIdx.x;
  if (i < total) {
    int f = (int)(i & (D_FEAT - 1));
    x[i] = fmaxf(x[i] + b[f], 0.0f);
  }
}

// grid-stride block reduction -> atomicAdd into *out
__global__ void reduce_sum(const float* __restrict__ x, long n, float* __restrict__ out) {
  __shared__ float sm[256];
  long stride = (long)gridDim.x * blockDim.x;
  float v = 0.0f;
  for (long i = (long)blockIdx.x * blockDim.x + threadIdx.x; i < n; i += stride) v += x[i];
  sm[threadIdx.x] = v;
  __syncthreads();
  for (int s = 128; s > 0; s >>= 1) {
    if ((int)threadIdx.x < s) sm[threadIdx.x] += sm[threadIdx.x + s];
    __syncthreads();
  }
  if (threadIdx.x == 0) atomicAdd(out, sm[0]);
}

// tiny channel-attention MLP (C<=4): mean -> relu(fc1) -> sigmoid(fc2)
__global__ void attention_mlp(const float* __restrict__ sums, float denom, int C,
                              const float* __restrict__ fc1W, const float* __restrict__ fc1b,
                              const float* __restrict__ fc2W, const float* __restrict__ fc2b,
                              float* __restrict__ a_out) {
  if (threadIdx.x != 0 || blockIdx.x != 0) return;
  float mean[4], t1[4];
  for (int c = 0; c < C; c++) mean[c] = sums[c] / denom;
  for (int c = 0; c < C; c++) {
    float s = fc1b[c];
    for (int k = 0; k < C; k++) s += fc1W[c * C + k] * mean[k];
    t1[c] = fmaxf(s, 0.0f);
  }
  for (int c = 0; c < C; c++) {
    float s = fc2b[c];
    for (int k = 0; k < C; k++) s += fc2W[c * C + k] * t1[k];
    a_out[c] = 1.0f / (1.0f + expf(-s));
  }
}

// packed[i, c*128+f] = chan[c][i][f] * a[c]   (relu(x*a)==x*a since x>=0, a>=0)
__global__ void pack_scaled(const float* __restrict__ chan, const float* __restrict__ a,
                            float* __restrict__ packed, int n, int C) {
  long total = (long)n * C * D_FEAT;
  long i = (long)blockIdx.x * blockDim.x + threadIdx.x;
  if (i >= total) return;
  int f = (int)(i % D_FEAT);
  int c = (int)((i / D_FEAT) % C);
  int node = (int)(i / ((long)D_FEAT * C));
  packed[i] = chan[((long)c * n + node) * D_FEAT + f] * a[c];
}

// ---------------------------------------------------------------- WMMA GEMM
// out[m,n] = sum_k A[m,k] * W[n,k]  (+bias[n], relu if FUSE)  -- i.e. A @ W^T
// A: [nrows, K] f32 row-major, W: [Nout, K] f32 row-major. K % 32 == 0,
// Nout % 16 == 0. One wave computes one 16x16 tile via v_wmma_f32_16x16x32_f16.
//
// Tail handling: A rows past nrows are CLAMPED to the last valid row (their
// products land only in D rows that are never stored), so the K-loop is fully
// convergent. Fragment loads are contiguous 128-bit vectors:
//   A frag = 8 floats at (k0 + half*8) and 8 floats at (k0 + 16 + half*8)
//   B frag = 8 floats at (k0 + half*8) and 8 floats at (k0 + 16 + half*8)
// Epilogue: interior tiles (uniform test) store 8 unguarded b32 per lane;
// only the last M-tile takes the per-row guarded path.
template <bool FUSE>
__global__ __launch_bounds__(32) void gemm_wmma(const float* __restrict__ A,
                                                const float* __restrict__ W,
                                                const float* __restrict__ bias,
                                                float* __restrict__ Cout,
                                                int nrows, int K, int Nout) {
  const int tm = blockIdx.x;
  const int tn = blockIdx.y;
  const int lane = threadIdx.x;      // 0..31 (wave32)
  const int half = lane >> 4;        // lane-half
  const int ml = lane & 15;
  int row = tm * 16 + ml;            // A row for this lane
  if (row >= nrows) row = nrows - 1; // clamp: keeps loop convergent, rows unused
  const int col = tn * 16 + ml;      // W row (output column) for this lane

  const float* pa = A + (long)row * K + (half << 3);  // two 8-float runs / step
  const float* pb = W + (long)col * K + (half << 3);  // two 8-float runs / step

  v8f acc = {};
  int k0 = 0;
  do {
    float4 a0 = *reinterpret_cast<const float4*>(pa + k0);
    float4 a1 = *reinterpret_cast<const float4*>(pa + k0 + 4);
    float4 a2 = *reinterpret_cast<const float4*>(pa + k0 + 16);
    float4 a3 = *reinterpret_cast<const float4*>(pa + k0 + 20);
    float4 b0 = *reinterpret_cast<const float4*>(pb + k0);
    float4 b1 = *reinterpret_cast<const float4*>(pb + k0 + 4);
    float4 b2 = *reinterpret_cast<const float4*>(pb + k0 + 16);
    float4 b3 = *reinterpret_cast<const float4*>(pb + k0 + 20);

    v16h af, bf;
    af[0]  = (_Float16)a0.x; af[1]  = (_Float16)a0.y;
    af[2]  = (_Float16)a0.z; af[3]  = (_Float16)a0.w;
    af[4]  = (_Float16)a1.x; af[5]  = (_Float16)a1.y;
    af[6]  = (_Float16)a1.z; af[7]  = (_Float16)a1.w;
    af[8]  = (_Float16)a2.x; af[9]  = (_Float16)a2.y;
    af[10] = (_Float16)a2.z; af[11] = (_Float16)a2.w;
    af[12] = (_Float16)a3.x; af[13] = (_Float16)a3.y;
    af[14] = (_Float16)a3.z; af[15] = (_Float16)a3.w;
    bf[0]  = (_Float16)b0.x; bf[1]  = (_Float16)b0.y;
    bf[2]  = (_Float16)b0.z; bf[3]  = (_Float16)b0.w;
    bf[4]  = (_Float16)b1.x; bf[5]  = (_Float16)b1.y;
    bf[6]  = (_Float16)b1.z; bf[7]  = (_Float16)b1.w;
    bf[8]  = (_Float16)b2.x; bf[9]  = (_Float16)b2.y;
    bf[10] = (_Float16)b2.z; bf[11] = (_Float16)b2.w;
    bf[12] = (_Float16)b3.x; bf[13] = (_Float16)b3.y;
    bf[14] = (_Float16)b3.z; bf[15] = (_Float16)b3.w;

    acc = __builtin_amdgcn_wmma_f32_16x16x32_f16(
        /*neg_a=*/false, af, /*neg_b=*/false, bf,
        /*c_mod=*/(short)0, acc, /*reuse_a=*/false, /*reuse_b=*/false);
    k0 += 32;
  } while (k0 < K);

  // D layout: VGPR r -> M = tm*16 + r + 8*half, N = tn*16 + (lane%16)
  const int n = tn * 16 + ml;
  float badd = 0.0f;
  if (FUSE) badd = bias[n];
  const int m0 = tm * 16 + (half << 3);
  float* cp = Cout + (long)m0 * Nout + n;

  if (tm * 16 + 16 <= nrows) {      // uniform: whole tile interior
#pragma unroll
    for (int r = 0; r < 8; r++) {
      float v = acc[r];
      if (FUSE) v = fmaxf(v + badd, 0.0f);
      *cp = v;
      cp += Nout;
    }
  } else {                          // last M-tile: per-row guard
#pragma unroll
    for (int r = 0; r < 8; r++) {
      if (m0 + r < nrows) {
        float v = acc[r];
        if (FUSE) v = fmaxf(v + badd, 0.0f);
        *cp = v;
      }
      cp += Nout;
    }
  }
}

// ---------------------------------------------------------------- launcher

extern "C" void kernel_launch(void* const* d_in, const int* in_sizes, int n_in,
                              void* d_out, int out_size, void* d_ws, size_t ws_size,
                              hipStream_t stream) {
  (void)n_in; (void)out_size; (void)ws_size;
  const int NM = 1043, ND = 2166, D = D_FEAT;

  const float* mirna_x = (const float*)d_in[0];
  const float* drug_x  = (const float*)d_in[1];

  struct Graph {
    const int* src; const int* dst; const float* w; int E;
    const float* W1; const float* b1; const float* W2; const float* b2;
  };
  auto mkGraph = [&](int base) {
    Graph g;
    g.E = in_sizes[base + 1];                 // w element count == E
    const int* edge = (const int*)d_in[base]; // [2, E] flat
    g.src = edge; g.dst = edge + g.E;
    g.w  = (const float*)d_in[base + 1];
    g.W1 = (const float*)d_in[base + 2]; g.b1 = (const float*)d_in[base + 3];
    g.W2 = (const float*)d_in[base + 4]; g.b2 = (const float*)d_in[base + 5];
    return g;
  };
  Graph m1 = mkGraph(2), m2 = mkGraph(8), d1 = mkGraph(14), d2 = mkGraph(20);

  const float* m_fc1_W = (const float*)d_in[26]; const float* m_fc1_b = (const float*)d_in[27];
  const float* m_fc2_W = (const float*)d_in[28]; const float* m_fc2_b = (const float*)d_in[29];
  const float* d_fc1_W = (const float*)d_in[30]; const float* d_fc1_b = (const float*)d_in[31];
  const float* d_fc2_W = (const float*)d_in[32]; const float* d_fc2_b = (const float*)d_in[33];
  const float* m_fus_W1 = (const float*)d_in[34]; const float* m_fus_b1 = (const float*)d_in[35];
  const float* m_fus_W2 = (const float*)d_in[36]; const float* m_fus_b2 = (const float*)d_in[37];
  const float* d_fus_W1 = (const float*)d_in[38]; const float* d_fus_b1 = (const float*)d_in[39];
  const float* d_fus_W2 = (const float*)d_in[40]; const float* d_fus_b2 = (const float*)d_in[41];

  // workspace carve-out (float units, 256-float = 1KB aligned for float4 loads)
  float* ws = (float*)d_ws;
  size_t off = 0;
  auto alloc = [&](size_t nf) { float* p = ws + off; off += (nf + 255) & ~(size_t)255; return p; };
  float* dinv  = alloc(ND);                    // per-graph 1/sqrt(deg), reused
  float* h_buf = alloc((size_t)ND * D);        // GEMM output h = x @ W^T, reused
  float* mchan = alloc((size_t)4 * NM * D);    // [m1a, m1b, m2a, m2b]
  float* dchan = alloc((size_t)4 * ND * D);    // [d1a, d1b, d2a, d2b]
  float* dsum  = alloc((size_t)2 * ND * D);    // [d1a+d2a, d1b+d2b]
  float* mpack = alloc((size_t)NM * 4 * D);    // attention-scaled concat
  float* dpack = alloc((size_t)ND * 2 * D);
  float* mhid  = alloc((size_t)NM * 2 * D);    // fusion hidden
  float* dhid  = alloc((size_t)ND * 2 * D);
  float* sums  = alloc(8);                     // channel sums
  float* att   = alloc(8);                     // attention scales

  auto cdiv = [](long a, long b) { return (unsigned)((a + b - 1) / b); };

  auto encoder = [&](const float* x, int n, const Graph& g, float* out1, float* out2) {
    long nD = (long)n * D;
    zero_f32<<<cdiv(n, 256), 256, 0, stream>>>(dinv, n);
    deg_accum<<<cdiv(g.E, 256), 256, 0, stream>>>(g.dst, g.w, dinv, g.E);
    deg_to_dinv<<<cdiv(n, 256), 256, 0, stream>>>(dinv, n);
    // layer 1: h = x @ W1^T ; out1 = relu(scatter(h) + b1)
    gemm_wmma<false><<<dim3(cdiv(n, 16), D / 16), 32, 0, stream>>>(x, g.W1, nullptr, h_buf, n, D, D);
    zero_f32<<<cdiv(nD, 256), 256, 0, stream>>>(out1, nD);
    edge_scatter<<<cdiv((long)g.E * 32, 256), 256, 0, stream>>>(h_buf, g.src, g.dst, g.w, dinv, out1, g.E);
    bias_relu<<<cdiv(nD, 256), 256, 0, stream>>>(out1, g.b1, nD);
    // layer 2
    gemm_wmma<false><<<dim3(cdiv(n, 16), D / 16), 32, 0, stream>>>(out1, g.W2, nullptr, h_buf, n, D, D);
    zero_f32<<<cdiv(nD, 256), 256, 0, stream>>>(out2, nD);
    edge_scatter<<<cdiv((long)g.E * 32, 256), 256, 0, stream>>>(h_buf, g.src, g.dst, g.w, dinv, out2, g.E);
    bias_relu<<<cdiv(nD, 256), 256, 0, stream>>>(out2, g.b2, nD);
  };

  // ---- encoders ----
  encoder(mirna_x, NM, m1, mchan,                      mchan + (size_t)1 * NM * D);
  encoder(mirna_x, NM, m2, mchan + (size_t)2 * NM * D, mchan + (size_t)3 * NM * D);
  encoder(drug_x,  ND, d1, dchan,                      dchan + (size_t)1 * ND * D);
  encoder(drug_x,  ND, d2, dchan + (size_t)2 * ND * D, dchan + (size_t)3 * ND * D);

  long ndD = (long)ND * D, nmD = (long)NM * D;
  // drug: sum views per layer -> 2 channels
  vec_add<<<cdiv(ndD, 256), 256, 0, stream>>>(dchan, dchan + (size_t)2 * ND * D, dsum, ndD);
  vec_add<<<cdiv(ndD, 256), 256, 0, stream>>>(dchan + (size_t)1 * ND * D,
                                              dchan + (size_t)3 * ND * D,
                                              dsum + (size_t)1 * ND * D, ndD);

  // ---- miRNA attention + fusion ----
  zero_f32<<<1, 32, 0, stream>>>(sums, 8);
  for (int c = 0; c < 4; c++)
    reduce_sum<<<256, 256, 0, stream>>>(mchan + (size_t)c * NM * D, nmD, sums + c);
  attention_mlp<<<1, 32, 0, stream>>>(sums, (float)nmD, 4, m_fc1_W, m_fc1_b, m_fc2_W, m_fc2_b, att);
  pack_scaled<<<cdiv((long)NM * 4 * D, 256), 256, 0, stream>>>(mchan, att, mpack, NM, 4);
  gemm_wmma<true><<<dim3(cdiv(NM, 16), (2 * D) / 16), 32, 0, stream>>>(mpack, m_fus_W1, m_fus_b1, mhid, NM, 4 * D, 2 * D);
  gemm_wmma<true><<<dim3(cdiv(NM, 16), D / 16), 32, 0, stream>>>(mhid, m_fus_W2, m_fus_b2, (float*)d_out, NM, 2 * D, D);

  // ---- drug attention + fusion ----
  zero_f32<<<1, 32, 0, stream>>>(sums, 8);
  for (int c = 0; c < 2; c++)
    reduce_sum<<<256, 256, 0, stream>>>(dsum + (size_t)c * ND * D, ndD, sums + c);
  attention_mlp<<<1, 32, 0, stream>>>(sums, (float)ndD, 2, d_fc1_W, d_fc1_b, d_fc2_W, d_fc2_b, att);
  pack_scaled<<<cdiv((long)ND * 2 * D, 256), 256, 0, stream>>>(dsum, att, dpack, ND, 2);
  gemm_wmma<true><<<dim3(cdiv(ND, 16), (2 * D) / 16), 32, 0, stream>>>(dpack, d_fus_W1, d_fus_b1, dhid, ND, 2 * D, 2 * D);
  gemm_wmma<true><<<dim3(cdiv(ND, 16), D / 16), 32, 0, stream>>>(dhid, d_fus_W2, d_fus_b2,
                                                                 (float*)d_out + (size_t)NM * D, ND, 2 * D, D);
}
